// MixtureOfExperts_6992206758377
// MI455X (gfx1250) — compile-verified, matrix operand order checked by reading, not run on hardware
//
#include <hip/hip_runtime.h>
#include <hip/hip_bf16.h>

// MoE: D=1024, F=4096, E=8, TOP_K=2, T = 2*2048 = 4096 tokens.
// fp32 router (exact top-2 softmax) + bf16 WMMA FFN (f32 accumulate),
// computed only for routed (token, expert) pairs.
// GEMMs: 8 waves/block, 32x32 tile per wave (4 x v_wmma_f32_16x16x32_bf16
// per K-step), A tile staged in LDS with double-buffered
// global_load_async_to_lds_b128 (ASYNCcnt-tracked).

#define D_ 1024
#define F_ 4096
#define E_ 8
#define T_ 4096

typedef __attribute__((ext_vector_type(16))) __bf16 v16bf;
typedef __attribute__((ext_vector_type(8)))  __bf16 v8bf;
typedef __attribute__((ext_vector_type(8)))  float  v8f;
typedef int v4i_m __attribute__((vector_size(16)));

// ---------------- workspace layout (bytes) ----------------
static constexpr size_t OFF_TOK = 1024;
static constexpr size_t OFF_WGT = OFF_TOK + (size_t)E_ * T_ * 4;
static constexpr size_t OFF_XB  = OFF_WGT + (size_t)E_ * T_ * 4;
static constexpr size_t OFF_W1T = OFF_XB  + (size_t)T_ * D_ * 2;
static constexpr size_t OFF_W2T = OFF_W1T + (size_t)E_ * D_ * F_ * 2;
static constexpr size_t OFF_H   = OFF_W2T + (size_t)E_ * D_ * F_ * 2;

// ---------------- CDNA5 async copy to LDS (guarded) ----------------
#if defined(__has_builtin)
#if __has_builtin(__builtin_amdgcn_global_load_async_to_lds_b128)
#define MOE_HAVE_ASYNC_LDS 1
#endif
#endif

__device__ __forceinline__ void stage16(const __bf16* g, __bf16* l) {
#ifdef MOE_HAVE_ASYNC_LDS
    __builtin_amdgcn_global_load_async_to_lds_b128(
        (__attribute__((address_space(1))) v4i_m*)g,
        (__attribute__((address_space(3))) v4i_m*)l, 0, 0);
#else
    *reinterpret_cast<v8bf*>(l) = *reinterpret_cast<const v8bf*>(g);
#endif
}

template <int N>
__device__ __forceinline__ void wait_async() {
#ifdef MOE_HAVE_ASYNC_LDS
#if __has_builtin(__builtin_amdgcn_s_wait_asynccnt)
    __builtin_amdgcn_s_wait_asynccnt(N);
#else
    asm volatile("s_wait_asynccnt %0" ::"n"(N) : "memory");
#endif
#endif
}

__device__ __forceinline__ float gelu_exact(float v) {
    return 0.5f * v * (1.0f + erff(v * 0.70710678118654752f));
}

// Load a 16x32 bf16 fragment: lane holds 8 contiguous K values at k-offset
// half*8 and 8 more at 16+half*8 (ISA 16-bit A/B layout). Works for global
// (global_load_b128) and LDS (ds_load_b128) pointers.
__device__ __forceinline__ v16bf ldfrag(const __bf16* base, int half) {
    const v8bf lo = *reinterpret_cast<const v8bf*>(base + half * 8);
    const v8bf hi = *reinterpret_cast<const v8bf*>(base + 16 + half * 8);
    v16bf r;
#pragma unroll
    for (int i = 0; i < 8; ++i) { r[i] = lo[i]; r[i + 8] = hi[i]; }
    return r;
}

__device__ __forceinline__ v8f wmma_bf16(v16bf a, v16bf b, v8f c) {
    return __builtin_amdgcn_wmma_f32_16x16x32_bf16(false, a, false, b, (short)0, c,
                                                   false, false);
}

// ---------------- small kernels ----------------

__global__ void moe_zero_kernel(float* __restrict__ out, int* __restrict__ counts, size_t n) {
    size_t g = (size_t)blockIdx.x * blockDim.x + threadIdx.x;
    if (g < n) out[g] = 0.0f;
    if (g < E_) counts[g] = 0;
}

__global__ void moe_cvt_x_kernel(const float* __restrict__ x, __bf16* __restrict__ xb, size_t n) {
    size_t g = (size_t)blockIdx.x * blockDim.x + threadIdx.x;
    if (g < n) xb[g] = (__bf16)x[g];
}

// Transpose-convert: src [E][R][C] fp32 -> dst [E][C][R] bf16. block (32,8).
__global__ void moe_transpose_kernel(const float* __restrict__ src, __bf16* __restrict__ dst,
                                     int R, int C) {
    __shared__ float tile[32][33];
    const int e = blockIdx.z;
    src += (size_t)e * R * C;
    dst += (size_t)e * R * C;
    const int c0 = blockIdx.x * 32;
    const int r0 = blockIdx.y * 32;
#pragma unroll
    for (int i = 0; i < 4; ++i) {
        int r = r0 + threadIdx.y + i * 8;
        tile[threadIdx.y + i * 8][threadIdx.x] = src[(size_t)r * C + c0 + threadIdx.x];
    }
    __syncthreads();
#pragma unroll
    for (int i = 0; i < 4; ++i) {
        int c = c0 + threadIdx.y + i * 8;
        dst[(size_t)c * R + r0 + threadIdx.x] = (__bf16)tile[threadIdx.x][threadIdx.y + i * 8];
    }
}

// One wave per token: logits = x . Wr + br, top-2 softmax, append to lists.
__global__ void moe_router_kernel(const float* __restrict__ x, const float* __restrict__ Wr,
                                  const float* __restrict__ br, int* __restrict__ counts,
                                  int* __restrict__ tok, float* __restrict__ wgt) {
    const int wave = threadIdx.x >> 5;
    const int lane = threadIdx.x & 31;
    const int t = blockIdx.x * 8 + wave;
    if (t >= T_) return;

    float acc[E_];
#pragma unroll
    for (int e = 0; e < E_; ++e) acc[e] = 0.0f;

    const float* xr = x + (size_t)t * D_;
    for (int d = lane; d < D_; d += 32) {
        float xv = xr[d];
        const float* wr = Wr + (size_t)d * E_;
#pragma unroll
        for (int e = 0; e < E_; ++e) acc[e] += xv * wr[e];
    }
#pragma unroll
    for (int e = 0; e < E_; ++e) {
#pragma unroll
        for (int off = 16; off > 0; off >>= 1) acc[e] += __shfl_xor(acc[e], off, 32);
    }
    if (lane == 0) {
        float lg[E_];
#pragma unroll
        for (int e = 0; e < E_; ++e) lg[e] = acc[e] + br[e];
        int i0 = 0;
#pragma unroll
        for (int e = 1; e < E_; ++e) if (lg[e] > lg[i0]) i0 = e;
        int i1 = (i0 == 0) ? 1 : 0;
#pragma unroll
        for (int e = 0; e < E_; ++e) if (e != i0 && lg[e] > lg[i1]) i1 = e;
        float m  = lg[i0];
        float e0 = expf(lg[i0] - m), e1 = expf(lg[i1] - m);
        float s  = e0 + e1;
        int p0 = atomicAdd(&counts[i0], 1);
        tok[i0 * T_ + p0] = t;  wgt[i0 * T_ + p0] = e0 / s;
        int p1 = atomicAdd(&counts[i1], 1);
        tok[i1 * T_ + p1] = t;  wgt[i1 * T_ + p1] = e1 / s;
    }
}

// ---------------- GEMM inner machinery ----------------
// One double-buffer phase: consume `buf` (A tile 32x64 in LDS) against two
// B panels, accumulating 4 WMMA tiles. Issues next-chunk stage before waiting.
struct Acc4 { v8f a00, a01, a10, a11; };

__device__ __forceinline__ void consume_chunk(const __bf16* buf, const __bf16* b0,
                                              const __bf16* b1, int mcol, int half,
                                              Acc4& acc) {
#pragma unroll
    for (int ks = 0; ks < 64; ks += 32) {
        v16bf A0 = ldfrag(&buf[mcol * 64 + ks], half);
        v16bf A1 = ldfrag(&buf[(mcol + 16) * 64 + ks], half);
        v16bf B0 = ldfrag(b0 + ks, half);
        v16bf B1 = ldfrag(b1 + ks, half);
        acc.a00 = wmma_bf16(A0, B0, acc.a00);
        acc.a01 = wmma_bf16(A0, B1, acc.a01);
        acc.a10 = wmma_bf16(A1, B0, acc.a10);
        acc.a11 = wmma_bf16(A1, B1, acc.a11);
    }
}

// Shared GEMM core: C[32 x 256 block] = A_gather[32 x K] * B[K x N].
// abuf: LDS double buffer. csrc: this thread's A-stage source row pointer.
template <int K>
__device__ __forceinline__ Acc4 gemm_core(const __bf16* csrc, __bf16* cdst0, __bf16* cdst1,
                                          const __bf16* brow0, const __bf16* brow1,
                                          __bf16* buf0, __bf16* buf1, int mcol, int half) {
    Acc4 acc;
    acc.a00 = {}; acc.a01 = {}; acc.a10 = {}; acc.a11 = {};
    constexpr int NCH = K / 64;   // even
    stage16(csrc, cdst0);
#pragma unroll 1
    for (int c = 0; c < NCH; c += 2) {
        // ---- phase 0: consume buf0, stage chunk c+1 into buf1 ----
        stage16(csrc + (c + 1) * 64, cdst1);
        wait_async<1>();
        __syncthreads();
        __builtin_prefetch(brow0 + (c + 1) * 64, 0, 1);
        __builtin_prefetch(brow1 + (c + 1) * 64, 0, 1);
        consume_chunk(buf0, brow0 + c * 64, brow1 + c * 64, mcol, half, acc);
        __syncthreads();
        // ---- phase 1: consume buf1, stage chunk c+2 into buf0 ----
        if (c + 2 < NCH) {
            stage16(csrc + (c + 2) * 64, cdst0);
            wait_async<1>();
        } else {
            wait_async<0>();
        }
        __syncthreads();
        __builtin_prefetch(brow0 + (c + 2) * 64, 0, 1);
        __builtin_prefetch(brow1 + (c + 2) * 64, 0, 1);
        consume_chunk(buf1, brow0 + (c + 1) * 64, brow1 + (c + 1) * 64, mcol, half, acc);
        __syncthreads();
    }
    return acc;
}

// ---------------- GEMM kernels ----------------
// Block: 256 thr = 8 waves. Block tile: M=32 rows, N=256 cols.
// grid.x = E * (T/32), grid.y = N_total/256.

__global__ void __launch_bounds__(256, 1)
moe_gemm1_kernel(const __bf16* __restrict__ xb, const __bf16* __restrict__ w1t,
                 const float* __restrict__ b1, const int* __restrict__ counts,
                 const int* __restrict__ tok, __bf16* __restrict__ h) {
    __shared__ __align__(16) __bf16 abuf[2][32 * 64];

    const int e  = blockIdx.x >> 7;      // 128 row tiles per expert
    const int rt = blockIdx.x & 127;
    const int cnt = counts[e];
    if (rt * 32 >= cnt) return;
    int hbase = 0;
    for (int i = 0; i < e; ++i) hbase += counts[i];

    const int tid  = threadIdx.x;
    const int wave = tid >> 5;
    const int lane = tid & 31;
    const int half = lane >> 4;
    const int mcol = lane & 15;
    const int n0   = blockIdx.y * 256 + wave * 32;

    // copy role: thread stages 16B (8 bf16) of the A tile per chunk
    const int crow  = tid >> 3;          // 0..31
    const int ckoff = (tid & 7) * 8;     // 0..56
    int crg = rt * 32 + crow;
    if (crg >= cnt) crg = cnt - 1;
    const __bf16* csrc = xb + (size_t)tok[e * T_ + crg] * D_ + ckoff;
    __bf16* cdst0 = &abuf[0][crow * 64 + ckoff];
    __bf16* cdst1 = &abuf[1][crow * 64 + ckoff];

    const __bf16* brow0 = w1t + ((size_t)e * F_ + (n0 + mcol)) * D_;
    const __bf16* brow1 = brow0 + (size_t)16 * D_;

    Acc4 acc = gemm_core<D_>(csrc, cdst0, cdst1, brow0, brow1,
                             abuf[0], abuf[1], mcol, half);

    const int f0 = n0 + mcol, f1 = n0 + mcol + 16;
    const float bb0 = b1[e * F_ + f0], bb1 = b1[e * F_ + f1];
#pragma unroll
    for (int j = 0; j < 8; ++j) {
        int r0 = rt * 32 + (half ? (j + 8) : j);
        int r1 = r0 + 16;
        if (r0 < cnt) {
            __bf16* hr = h + (size_t)(hbase + r0) * F_;
            hr[f0] = (__bf16)gelu_exact(acc.a00[j] + bb0);
            hr[f1] = (__bf16)gelu_exact(acc.a01[j] + bb1);
        }
        if (r1 < cnt) {
            __bf16* hr = h + (size_t)(hbase + r1) * F_;
            hr[f0] = (__bf16)gelu_exact(acc.a10[j] + bb0);
            hr[f1] = (__bf16)gelu_exact(acc.a11[j] + bb1);
        }
    }
}

__global__ void __launch_bounds__(256, 1)
moe_gemm2_kernel(const __bf16* __restrict__ h, const __bf16* __restrict__ w2t,
                 const float* __restrict__ b2, const int* __restrict__ counts,
                 const int* __restrict__ tok, const float* __restrict__ wgt,
                 float* __restrict__ out) {
    __shared__ __align__(16) __bf16 abuf[2][32 * 64];

    const int e  = blockIdx.x >> 7;
    const int rt = blockIdx.x & 127;
    const int cnt = counts[e];
    if (rt * 32 >= cnt) return;
    int hbase = 0;
    for (int i = 0; i < e; ++i) hbase += counts[i];

    const int tid  = threadIdx.x;
    const int wave = tid >> 5;
    const int lane = tid & 31;
    const int half = lane >> 4;
    const int mcol = lane & 15;
    const int n0   = blockIdx.y * 256 + wave * 32;

    const int crow  = tid >> 3;
    const int ckoff = (tid & 7) * 8;
    int crg = rt * 32 + crow;
    if (crg >= cnt) crg = cnt - 1;
    const __bf16* csrc = h + (size_t)(hbase + crg) * F_ + ckoff;
    __bf16* cdst0 = &abuf[0][crow * 64 + ckoff];
    __bf16* cdst1 = &abuf[1][crow * 64 + ckoff];

    const __bf16* brow0 = w2t + ((size_t)e * D_ + (n0 + mcol)) * F_;
    const __bf16* brow1 = brow0 + (size_t)16 * F_;

    Acc4 acc = gemm_core<F_>(csrc, cdst0, cdst1, brow0, brow1,
                             abuf[0], abuf[1], mcol, half);

    const int d0 = n0 + mcol, d1 = n0 + mcol + 16;
    const float bb0 = b2[e * D_ + d0], bb1 = b2[e * D_ + d1];
#pragma unroll
    for (int j = 0; j < 8; ++j) {
        int r0 = rt * 32 + (half ? (j + 8) : j);
        int r1 = r0 + 16;
        if (r0 < cnt) {
            const int tk = tok[e * T_ + r0];
            const float w = wgt[e * T_ + r0];
            unsafeAtomicAdd(&out[(size_t)tk * D_ + d0], (acc.a00[j] + bb0) * w);
            unsafeAtomicAdd(&out[(size_t)tk * D_ + d1], (acc.a01[j] + bb1) * w);
        }
        if (r1 < cnt) {
            const int tk = tok[e * T_ + r1];
            const float w = wgt[e * T_ + r1];
            unsafeAtomicAdd(&out[(size_t)tk * D_ + d0], (acc.a10[j] + bb0) * w);
            unsafeAtomicAdd(&out[(size_t)tk * D_ + d1], (acc.a11[j] + bb1) * w);
        }
    }
}

// ---------------- host launcher ----------------
extern "C" void kernel_launch(void* const* d_in, const int* in_sizes, int n_in,
                              void* d_out, int out_size, void* d_ws, size_t ws_size,
                              hipStream_t stream) {
    const float* x  = (const float*)d_in[0];
    const float* Wr = (const float*)d_in[1];
    const float* br = (const float*)d_in[2];
    const float* W1 = (const float*)d_in[3];
    const float* b1 = (const float*)d_in[4];
    const float* W2 = (const float*)d_in[5];
    const float* b2 = (const float*)d_in[6];
    float* out = (float*)d_out;

    char* ws = (char*)d_ws;
    int*    counts = (int*)ws;
    int*    tok    = (int*)(ws + OFF_TOK);
    float*  wgt    = (float*)(ws + OFF_WGT);
    __bf16* xb     = (__bf16*)(ws + OFF_XB);
    __bf16* w1t    = (__bf16*)(ws + OFF_W1T);
    __bf16* w2t    = (__bf16*)(ws + OFF_W2T);
    __bf16* hbuf   = (__bf16*)(ws + OFF_H);

    const size_t nOut = (size_t)T_ * D_;

    moe_zero_kernel<<<dim3((unsigned)((nOut + 255) / 256)), dim3(256), 0, stream>>>(out, counts, nOut);
    moe_cvt_x_kernel<<<dim3((unsigned)((nOut + 255) / 256)), dim3(256), 0, stream>>>(x, xb, nOut);

    // W1 [E][D][F] -> w1t [E][F][D]
    moe_transpose_kernel<<<dim3(F_ / 32, D_ / 32, E_), dim3(32, 8), 0, stream>>>(W1, w1t, D_, F_);
    // W2 [E][F][D] -> w2t [E][D][F]
    moe_transpose_kernel<<<dim3(D_ / 32, F_ / 32, E_), dim3(32, 8), 0, stream>>>(W2, w2t, F_, D_);

    moe_router_kernel<<<dim3(T_ / 8), dim3(256), 0, stream>>>(x, Wr, br, counts, tok, wgt);

    moe_gemm1_kernel<<<dim3(E_ * (T_ / 32), F_ / 256), dim3(256), 0, stream>>>(
        xb, w1t, b1, counts, tok, hbuf);
    moe_gemm2_kernel<<<dim3(E_ * (T_ / 32), D_ / 256), dim3(256), 0, stream>>>(
        hbuf, w2t, b2, counts, tok, wgt, out);
}
